// PointConvolution_5531917877551
// MI455X (gfx1250) — matrix operand order repaired and checked by reading, not compiled, not run on hardware
//
#include <hip/hip_runtime.h>

// out = conv5x5(x, K) + b where K folds the 24 shift-subtract taps and the
// 1x1 channel mix: K[o][k!=12] = W[o][c(k)], K[o][12] = -sum_c W[o][c].
// GEMM per 16-pixel tile: D(32ch x 16px) = K(32x32pad) x P(32pad x 16px),
// computed as bf16x3-split v_wmma_f32_16x16x32_bf16 with f32 accumulate.
// Output-write bound: 512MB @ 23.3TB/s ~= 22us floor.

typedef __bf16 bf16_t;
typedef __attribute__((ext_vector_type(16))) __bf16 v16bf;
typedef __attribute__((ext_vector_type(8)))  float  v8f;

#define IMG_W 512
#define CH_STRIDE (512 * 512)

static __device__ __forceinline__ v8f wmma_bf16(v16bf a, v16bf b, v8f c) {
  // (neg_a, A, neg_b, B, c_mod, C, reuse_a, reuse_b)
  return __builtin_amdgcn_wmma_f32_16x16x32_bf16(false, a, false, b,
                                                 (short)0, c, false, false);
}

static __device__ __forceinline__ int clamp511(int v) {
  return v < 0 ? 0 : (v > 511 ? 511 : v);
}

__global__ __launch_bounds__(256) void pconv5x5_wmma_kernel(
    const float* __restrict__ x,     // [16,1,512,512]
    const float* __restrict__ W,     // [32,24]
    const float* __restrict__ bvec,  // [32]
    float* __restrict__ out)         // [16,32,512,512]
{
  const int lane  = threadIdx.x & 31;
  const int wv    = threadIdx.x >> 5;
  const int g     = blockIdx.x * 8 + wv;   // global row id, 0..8191
  const int bimg  = g >> 9;                // image index
  const int y     = g & 511;               // row in image
  const int m     = lane & 15;             // A: channel row / B,D: pixel col
  const int khalf = lane >> 4;

  // ---------------- A fragments: fold 1x1-mix into 5x5 kernel, bf16 hi/lo ----
  // A layout (contiguous k order, matched by B): lane m holds K {0..15}; lane m+16 holds
  // K {16..31}. element e -> K = 16*khalf + e.
  v16bf ah0, al0, ah1, al1;
  #pragma unroll
  for (int t = 0; t < 2; ++t) {
    const int o = t * 16 + m;
    const float* wr = W + o * 24;
    float wsum = 0.f;
    #pragma unroll
    for (int c = 0; c < 24; ++c) wsum += wr[c];
    #pragma unroll
    for (int e = 0; e < 16; ++e) {
      const int K = khalf * 16 + e;
      float v;
      if (K >= 25)      v = 0.f;        // K padding 25..31
      else if (K == 12) v = -wsum;      // center tap = -sum(W row)
      else              v = wr[K < 12 ? K : K - 1];
      const bf16_t h = (bf16_t)v;
      const bf16_t l = (bf16_t)(v - (float)h);
      if (t == 0) { ah0[e] = h; al0[e] = l; }
      else        { ah1[e] = h; al1[e] = l; }
    }
  }

  // ---------------- bias -> accumulator init (D: VGPR r = channel r+8*khalf) --
  v8f bias0, bias1;
  {
    const float* bp = bvec + khalf * 8;
    #pragma unroll
    for (int r = 0; r < 8; ++r) { bias0[r] = bp[r]; bias1[r] = bp[16 + r]; }
  }

  // ---------------- per-e neighborhood offsets (B: lane n holds K 16*khalf+e)
  const float* img = x + (size_t)bimg * CH_STRIDE;
  int offE[16];  // clamped_row_base + (dx-2): add pixel column -> elem offset
  int okE[16];   // valid (K<25 and row in range)
  #pragma unroll
  for (int e = 0; e < 16; ++e) {
    const int k  = khalf * 16 + e;
    const int dy = k / 5;
    const int dx = k - dy * 5;
    const int ry = y + dy - 2;
    okE[e]  = (k < 25) & (ry >= 0) & (ry < 512);
    offE[e] = clamp511(ry) * IMG_W + (dx - 2);
  }

  float* obase = out + ((size_t)(bimg * 32 + khalf * 8) * IMG_W + y) * IMG_W + m;

  // -------- one tile = 16 pixels x 32 channels; `edge` is constant-folded ----
  auto do_tile = [&](int tile, bool edge) {
    const int x0   = tile * 16;
    const int xcol = x0 + m;

    float v[16];
    if (edge) {
      // edge tiles (x0 = 0 or 496): clamp columns too
      #pragma unroll
      for (int e = 0; e < 16; ++e) {
        const int k  = khalf * 16 + e;
        const int dy = k / 5;
        const int dx = k - dy * 5;
        const int ry = y + dy - 2;
        const int rx = xcol + dx - 2;
        const float val = img[clamp511(ry) * IMG_W + clamp511(rx)];
        v[e] = (okE[e] & (rx >= 0) & (rx < 512)) ? val : 0.f;
      }
    } else {
      // interior: columns always in range, branch-free
      #pragma unroll
      for (int e = 0; e < 16; ++e) {
        const float val = img[offE[e] + xcol];
        v[e] = okE[e] ? val : 0.f;
      }
    }

    // bf16 hi/lo split of the B fragment
    v16bf bh, bl;
    #pragma unroll
    for (int e = 0; e < 16; ++e) {
      const bf16_t h = (bf16_t)v[e];
      bh[e] = h;
      bl[e] = (bf16_t)(v[e] - (float)h);
    }

    // bf16x3: Ahi*Bhi + Ahi*Blo + Alo*Bhi, f32 accumulate seeded with bias
    v8f acc0 = bias0, acc1 = bias1;
    acc0 = wmma_bf16(ah0, bh, acc0);
    acc1 = wmma_bf16(ah1, bh, acc1);
    acc0 = wmma_bf16(ah0, bl, acc0);
    acc1 = wmma_bf16(ah1, bl, acc1);
    acc0 = wmma_bf16(al0, bh, acc0);
    acc1 = wmma_bf16(al1, bh, acc1);

    // D: VGPR r = channel (8*khalf + r [+16]), lanes = 16 contiguous pixels
    float* op = obase + x0;
    #pragma unroll
    for (int r = 0; r < 8; ++r) {
      op[(size_t)r * CH_STRIDE]        = acc0[r];
      op[(size_t)(r + 16) * CH_STRIDE] = acc1[r];
    }
  };

  // peeled edges -> tight branch-free interior loop (30 of 32 tiles)
  do_tile(0, true);
  #pragma unroll 1
  for (int tile = 1; tile < 31; ++tile) do_tile(tile, false);
  do_tile(31, true);
}

extern "C" void kernel_launch(void* const* d_in, const int* in_sizes, int n_in,
                              void* d_out, int out_size, void* d_ws, size_t ws_size,
                              hipStream_t stream) {
  const float* x = (const float*)d_in[0];  // [16,1,512,512] f32
  const float* W = (const float*)d_in[1];  // [32,24] f32
  const float* b = (const float*)d_in[2];  // [32] f32
  float* out = (float*)d_out;              // [16,32,512,512] f32

  // 8 waves/block, one image-row per wave: 16*512 rows / 8 = 1024 blocks
  pconv5x5_wmma_kernel<<<dim3(1024), dim3(256), 0, stream>>>(x, W, b, out);
}
